// BMLP_33311766347902
// MI455X (gfx1250) — compile-verified, matrix operand order checked by reading, not run on hardware
//
#include <hip/hip_runtime.h>

// Binarized 3-layer MLP for MI455X (gfx1250, wave32).
// All GEMM operands are +/-1 -> V_WMMA_I32_16X16X64_IU8 (exact int32 accum).
// Operands pre-packed into the WMMA 8-bit fragment layout (contiguous 32B/lane).
// GEMM: 256 threads = 8 waves, block tile 256M x 64N; each wave owns 2 M-tiles
// x 4 N-tiles (8 accumulators) so every LDS B fragment feeds 2 WMMAs.
// Double-buffered LDS B staging -> 1 barrier per K-step, global B load
// overlapped with WMMA. Epilogue fuses bias+BN and writes the next layer's
// packed +/-1 input directly (no separate binarize pass).

typedef __attribute__((ext_vector_type(8))) int v8i;
typedef __attribute__((ext_vector_type(4))) int v4i;

#define QTHRESH (-1.0f / 256.0f)  // sign(quantize8(x)) >= 0  <=>  x >= -1/256

// ---------------------------------------------------------------------------
// Pack +/-1 bytes into WMMA 8-bit fragment layout. Tile = 16 rows x 64 K.
// (ISA 7.12.2): lane = (row%16) + 16*hi, hi = (k%16 >= 8);
// vgpr v, byte b: k_local = 32*(v>>2) + 16*((v>>1)&1) + 4*(v&1) + b + 8*hi.
// Storage: tile_base + lane*32 + v*4 + b, tiles row-major (tile = rt*KT + kt).
// ---------------------------------------------------------------------------
__global__ __launch_bounds__(256) void pack_sign(
    const float* __restrict__ src, signed char* __restrict__ dst,
    int K, int rows, int KT, float thresh)
{
    const int tid  = threadIdx.x;
    const int lane = tid >> 3;   // 0..31
    const int v    = tid & 7;    // 0..7
    const int rt   = blockIdx.x;
    const int kt   = blockIdx.y;
    const int r    = rt * 16 + (lane & 15);
    const int k0   = kt * 64 + ((lane >> 4) << 3)
                   + ((v >> 2) << 5) + (((v >> 1) & 1) << 4) + ((v & 1) << 2);

    unsigned w = 0x01010101u;  // padded rows -> +1 (results masked later)
    if (r < rows) {
        const float4 f = *(const float4*)(src + (size_t)r * K + k0);
        const unsigned b0 = (unsigned char)(signed char)((f.x >= thresh) ? 1 : -1);
        const unsigned b1 = (unsigned char)(signed char)((f.y >= thresh) ? 1 : -1);
        const unsigned b2 = (unsigned char)(signed char)((f.z >= thresh) ? 1 : -1);
        const unsigned b3 = (unsigned char)(signed char)((f.w >= thresh) ? 1 : -1);
        w = b0 | (b1 << 8) | (b2 << 16) | (b3 << 24);
    }
    *(unsigned*)(dst + ((size_t)(rt * KT + kt)) * 1024 + lane * 32 + v * 4) = w;
}

// ---------------------------------------------------------------------------
// Binary GEMM with fused BN epilogue.
// mode 0: binarize, scatter into next layer's packed A layout.
// mode 1: f32 output, columns >= Nreal dropped.
// ---------------------------------------------------------------------------
__global__ __launch_bounds__(256) void bgemm_bnn(
    const signed char* __restrict__ Ap,
    const signed char* __restrict__ Bp,
    int KT,
    const float* __restrict__ bias,  const float* __restrict__ gamma,
    const float* __restrict__ beta,  const float* __restrict__ mean,
    const float* __restrict__ var,
    int mode,
    signed char* __restrict__ outPacked, int outKT,
    float* __restrict__ outF, int Nreal)
{
    __shared__ alignas(64) signed char ldsB[2][4096];

    const int tid    = threadIdx.x;
    const int wave   = tid >> 5;
    const int lane   = tid & 31;
    const int mtile0 = blockIdx.x * 16 + wave * 2;  // 2 M-tiles per wave
    const int ntile0 = blockIdx.y * 4;              // 4 N-tiles per block

    const v8i zero = {0, 0, 0, 0, 0, 0, 0, 0};
    v8i acc[2][4] = {{zero, zero, zero, zero}, {zero, zero, zero, zero}};

    // B staging: thread t copies 16B of fragment (ntile0 + t/64), chunk t%64.
    const int bj = tid >> 6;
    const int bw = tid & 63;
    const signed char* bsrc = Bp + ((size_t)(ntile0 + bj) * KT) * 1024 + bw * 16;
    const signed char* a0   = Ap + ((size_t)mtile0 * KT) * 1024 + lane * 32;
    const signed char* a1   = a0 + (size_t)KT * 1024;

    // Prologue: stage B(0) into buffer 0.
    *(v4i*)(ldsB[0] + tid * 16) = *(const v4i*)bsrc;
    __syncthreads();

    for (int kt = 0; kt < KT; ++kt) {
        const bool more = (kt + 1 < KT);  // uniform; EXEC stays all-ones
        v4i breg = {0, 0, 0, 0};
        if (more) breg = *(const v4i*)(bsrc + (size_t)(kt + 1) * 1024);

        const v8i av0 = *(const v8i*)(a0 + (size_t)kt * 1024);
        const v8i av1 = *(const v8i*)(a1 + (size_t)kt * 1024);
        if (more) {
            __builtin_prefetch(a0 + (size_t)(kt + 1) * 1024, 0, 1);
            __builtin_prefetch(a1 + (size_t)(kt + 1) * 1024, 0, 1);
        }

        const signed char* lb = ldsB[kt & 1];
#pragma unroll
        for (int j = 0; j < 4; ++j) {
            const v8i bv = *(const v8i*)(lb + j * 1024 + lane * 32);
            acc[0][j] = __builtin_amdgcn_wmma_i32_16x16x64_iu8(
                true, av0, true, bv, acc[0][j], false, false);
            acc[1][j] = __builtin_amdgcn_wmma_i32_16x16x64_iu8(
                true, av1, true, bv, acc[1][j], false, false);
        }

        // Stage next B into the other buffer; single barrier per K-step:
        //  - my write(kt->buf^1) vs others' read(kt+1->buf^1): split by barrier
        //  - others' read(kt->buf) vs my write(kt+1->buf): split by barrier
        if (more) *(v4i*)(ldsB[(kt + 1) & 1] + tid * 16) = breg;
        __syncthreads();
    }

    // Epilogue. C layout: lane L, vgpr r -> (m_local = r + 8*(L>=16), n = L%16).
#pragma unroll
    for (int j = 0; j < 4; ++j) {
        const int   n  = (ntile0 + j) * 16 + (lane & 15);
        const float s  = gamma[n] * rsqrtf(var[n] + 1e-5f);
        const float c0 = bias[n] - mean[n];
        const float e  = beta[n];
        const int   mh = (lane >> 4) << 3;

        if (mode == 0) {
            // next-layer A element (m, k = n): invert fragment layout
            const int kt2 = n >> 6;
            const int kl  = n & 63;
            const int hi  = (kl >> 3) & 1;
            const int k2  = kl - (hi << 3);
            const int vv  = ((k2 >> 5) << 2) | (((k2 >> 4) & 1) << 1) | ((k2 >> 2) & 1);
            const int bb  = k2 & 3;
#pragma unroll
            for (int mi = 0; mi < 2; ++mi) {
                signed char* base = outPacked
                    + ((size_t)((mtile0 + mi) * outKT + kt2)) * 1024 + vv * 4 + bb;
#pragma unroll
                for (int r = 0; r < 8; ++r) {
                    const float o = ((float)acc[mi][j][r] + c0) * s + e;
                    base[((r + mh) + (hi << 4)) * 32] =
                        (o >= 0.0f) ? (signed char)1 : (signed char)-1;
                }
            }
        } else if (n < Nreal) {
#pragma unroll
            for (int mi = 0; mi < 2; ++mi) {
                const int mg = (mtile0 + mi) * 16 + mh;
#pragma unroll
                for (int r = 0; r < 8; ++r) {
                    const float o = ((float)acc[mi][j][r] + c0) * s + e;
                    outF[(size_t)(mg + r) * Nreal + n] = o;
                }
            }
        }
    }
}

// ---------------------------------------------------------------------------
// Inputs (setup_inputs order):
//  0:x  1:W1 2:b1 3:g1 4:be1 5:m1 6:v1  7:W2 ... 12:v2  13:W3 ... 18:v3
// ---------------------------------------------------------------------------
extern "C" void kernel_launch(void* const* d_in, const int* in_sizes, int n_in,
                              void* d_out, int out_size, void* d_ws, size_t ws_size,
                              hipStream_t stream)
{
    (void)in_sizes; (void)n_in; (void)out_size; (void)ws_size;

    const float* x   = (const float*)d_in[0];
    const float* W1  = (const float*)d_in[1];
    const float* b1  = (const float*)d_in[2];
    const float* g1  = (const float*)d_in[3];
    const float* be1 = (const float*)d_in[4];
    const float* m1  = (const float*)d_in[5];
    const float* v1  = (const float*)d_in[6];
    const float* W2  = (const float*)d_in[7];
    const float* b2  = (const float*)d_in[8];
    const float* g2  = (const float*)d_in[9];
    const float* be2 = (const float*)d_in[10];
    const float* m2  = (const float*)d_in[11];
    const float* v2  = (const float*)d_in[12];
    const float* W3  = (const float*)d_in[13];
    const float* b3  = (const float*)d_in[14];
    const float* g3  = (const float*)d_in[15];
    const float* be3 = (const float*)d_in[16];
    const float* m3  = (const float*)d_in[17];
    const float* v3  = (const float*)d_in[18];

    const int B = 8192, D = 4096, KT = 64;   // KT = 4096/64 K-tiles
    const int NOUT = 1000, N3PAD = 1024;

    signed char* ws  = (signed char*)d_ws;
    const size_t MiB = (size_t)1 << 20;
    signed char* Xp  = ws;             // 32 MiB packed activations (reused as H2)
    signed char* H1  = ws + 32 * MiB;  // 32 MiB
    signed char* W1p = ws + 64 * MiB;  // 16 MiB
    signed char* W2p = ws + 80 * MiB;  // 16 MiB
    signed char* W3p = ws + 96 * MiB;  //  4 MiB (1024x4096)

    // Binarize + pack into WMMA fragment layout
    pack_sign<<<dim3(B / 16,     KT), 256, 0, stream>>>(x,  Xp,  D, B,    KT, QTHRESH);
    pack_sign<<<dim3(D / 16,     KT), 256, 0, stream>>>(W1, W1p, D, D,    KT, 0.0f);
    pack_sign<<<dim3(D / 16,     KT), 256, 0, stream>>>(W2, W2p, D, D,    KT, 0.0f);
    pack_sign<<<dim3(N3PAD / 16, KT), 256, 0, stream>>>(W3, W3p, D, NOUT, KT, 0.0f);

    // Layer 1: (8192x4096) @ (4096x4096)^T -> binarized packed H1
    bgemm_bnn<<<dim3(B / 256, D / 64), 256, 0, stream>>>(
        Xp, W1p, KT, b1, g1, be1, m1, v1, 0, H1, KT, nullptr, 0);
    // Layer 2: -> binarized packed (reuse Xp)
    bgemm_bnn<<<dim3(B / 256, D / 64), 256, 0, stream>>>(
        H1, W2p, KT, b2, g2, be2, m2, v2, 0, Xp, KT, nullptr, 0);
    // Layer 3: f32 output 8192x1000
    bgemm_bnn<<<dim3(B / 256, N3PAD / 64), 256, 0, stream>>>(
        Xp, W3p, KT, b3, g3, be3, m3, v3, 1, nullptr, 0, (float*)d_out, NOUT);
}
